// Cell_52793738003180
// MI455X (gfx1250) — compile-verified, matrix operand order checked by reading, not run on hardware
//
#include <hip/hip_runtime.h>
#include <hip/hip_bf16.h>

// ---------------- problem constants ----------------
constexpr int   DIM    = 512;
constexpr int   NNODES = 50000;
constexpr int   NEDGES = 800000;
constexpr float EPSV   = 1e-5f;

typedef __attribute__((ext_vector_type(16))) __bf16 v16bf;
typedef __attribute__((ext_vector_type(8)))  __bf16 v8bf;
typedef __attribute__((ext_vector_type(8)))  float  v8f;

// ---------------- generic f32 -> bf16 plane copy ----------------
__global__ void cvt_bf16_kernel(const float* __restrict__ X, __bf16* __restrict__ Y,
                                long long n) {
  long long i = (long long)blockIdx.x * blockDim.x + threadIdx.x;
  if (i < n) Y[i] = (__bf16)X[i];
}

// ---------------- aggregate: init ----------------
__global__ void agg_init_kernel(float* __restrict__ s, int* __restrict__ mxbits,
                                float* __restrict__ deg, int n) {
  long long i = (long long)blockIdx.x * blockDim.x + threadIdx.x;
  long long tot = (long long)n * DIM;
  if (i < tot) { s[i] = 0.0f; mxbits[i] = 0xFF800000; } // -inf bits
  if (i < n) deg[i] = 0.0f;
}

// float atomic max via sign-split int trick (valid for all orderings)
__device__ __forceinline__ void atomic_fmax(int* p, float v) {
  if (v >= 0.0f) atomicMax(p, __float_as_int(v));
  else           atomicMin((unsigned int*)p, __float_as_uint(v));
}

// ---------------- aggregate: edge scatter (sum, max, degree) ----------------
__global__ void scatter_kernel(const float* __restrict__ X, const int* __restrict__ src,
                               const int* __restrict__ dst, float* __restrict__ s,
                               int* __restrict__ mxbits, float* __restrict__ deg) {
  long long t = (long long)blockIdx.x * blockDim.x + threadIdx.x;
  const long long tot = (long long)NEDGES * (DIM / 4);
  if (t >= tot) return;
  const int e = (int)(t >> 7);        // DIM/4 == 128 chunks per edge
  const int c = ((int)t & 127) << 2;  // starting channel
  const int si = src[e], di = dst[e];
  const float4 v = *(const float4*)(X + (long long)si * DIM + c);
  float* sp = s      + (long long)di * DIM + c;
  int*   mp = mxbits + (long long)di * DIM + c;
  atomicAdd(sp + 0, v.x); atomicAdd(sp + 1, v.y);
  atomicAdd(sp + 2, v.z); atomicAdd(sp + 3, v.w);
  atomic_fmax(mp + 0, v.x); atomic_fmax(mp + 1, v.y);
  atomic_fmax(mp + 2, v.z); atomic_fmax(mp + 3, v.w);
  if (c == 0) atomicAdd(deg + di, 1.0f);
}

// ---------------- aggregate: finalize -> bf16 planes (sum, mean, max) ----------------
__global__ void agg_fin_kernel(const float* __restrict__ s, const int* __restrict__ mxbits,
                               const float* __restrict__ deg, __bf16* __restrict__ sB,
                               __bf16* __restrict__ mB, __bf16* __restrict__ xB, int n) {
  long long i = (long long)blockIdx.x * blockDim.x + threadIdx.x;
  if (i >= (long long)n * DIM) return;
  const float dg  = deg[i >> 9];               // DIM == 512
  const float inv = 1.0f / fmaxf(dg, 1.0f);
  const float sv  = s[i];
  sB[i] = (__bf16)sv;
  mB[i] = (__bf16)(sv * inv);
  xB[i] = (__bf16)((dg > 0.0f) ? __int_as_float(mxbits[i]) : 0.0f);
}

// ---------------- W (f32 [k][d][e]) -> Wt (bf16 [k][e][d]) ----------------
__global__ void wconv_kernel(const float* __restrict__ W, __bf16* __restrict__ Wt, int nmat) {
  long long i = (long long)blockIdx.x * blockDim.x + threadIdx.x;
  if (i >= (long long)nmat * DIM * DIM) return;
  const int k   = (int)(i >> 18);          // DIM*DIM == 2^18
  const int rem = (int)i & (DIM * DIM - 1);
  const int d   = rem >> 9;
  const int e   = rem & (DIM - 1);
  Wt[((long long)k << 18) + ((long long)e << 9) + d] = (__bf16)W[i];
}

// pack two contiguous 8xbf16 chunks into the 16x16x32 fragment order
__device__ __forceinline__ v16bf frag16(const __bf16* row, int off0, int off1) {
  const v8bf lo = *(const v8bf*)(row + off0);
  const v8bf hi = *(const v8bf*)(row + off1);
  return __builtin_shufflevector(lo, hi, 0, 1, 2, 3, 4, 5, 6, 7,
                                 8, 9, 10, 11, 12, 13, 14, 15);
}

// ---------------- WMMA bf16 GEMM: Z[M x 512] (+)= A[M x 512] * Bt^T ----------------
// A, Bt: bf16 row-major with K=512 rows of length 512 (Bt[n][k] = B[k][n]).
// One wave computes a 16x64 strip: 4 independent accumulators share one A fragment.
__global__ void __launch_bounds__(128) gemm_kernel(const __bf16* __restrict__ A,
                                                   const __bf16* __restrict__ Bt,
                                                   float* __restrict__ Z,
                                                   int M, int addZ) {
  const int lane  = threadIdx.x & 31;
  const int wave  = threadIdx.x >> 5;
  const int r     = lane & 15;
  const int hi    = lane >> 4;
  const int tileM = blockIdx.x * 16;
  const int tileN = (blockIdx.y * 4 + wave) * 64;     // 4 N-tiles per wave
  const __bf16* arow = A  + (long long)(tileM + r) * DIM;
  const __bf16* brow = Bt + (long long)(tileN + r) * DIM;
  const int off0 = hi ? 8 : 0;    // 16-bit A layout: lanes<16 take K 0-7 & 16-23,
  const int off1 = hi ? 24 : 16;  //                  lanes>=16 take K 8-15 & 24-31
  v8f c0 = {}, c1 = {}, c2 = {}, c3 = {};
#pragma unroll 2
  for (int kb = 0; kb < DIM; kb += 32) {
    const v16bf af  = frag16(arow, kb + off0, kb + off1);
    const v16bf bf0 = frag16(brow + 0 * 16 * DIM, kb + off0, kb + off1);
    const v16bf bf1 = frag16(brow + 1 * 16 * DIM, kb + off0, kb + off1);
    const v16bf bf2 = frag16(brow + 2 * 16 * DIM, kb + off0, kb + off1);
    const v16bf bf3 = frag16(brow + 3 * 16 * DIM, kb + off0, kb + off1);
    c0 = __builtin_amdgcn_wmma_f32_16x16x32_bf16(false, af, false, bf0, (short)0, c0, false, false);
    c1 = __builtin_amdgcn_wmma_f32_16x16x32_bf16(false, af, false, bf1, (short)0, c1, false, false);
    c2 = __builtin_amdgcn_wmma_f32_16x16x32_bf16(false, af, false, bf2, (short)0, c2, false, false);
    c3 = __builtin_amdgcn_wmma_f32_16x16x32_bf16(false, af, false, bf3, (short)0, c3, false, false);
  }
  // C/D layout: VGPR v -> row v + 8*hi, col = lane&15
#pragma unroll
  for (int v = 0; v < 8; ++v) {
    float* p = Z + (long long)(tileM + v + hi * 8) * DIM + tileN + r;
    if (addZ) {
      p[0] += c0[v]; p[16] += c1[v]; p[32] += c2[v]; p[48] += c3[v];
    } else {
      p[0] = c0[v]; p[16] = c1[v]; p[32] = c2[v]; p[48] = c3[v];
    }
  }
}

// ---------------- small zero ----------------
__global__ void zero_kernel(float* __restrict__ p, int n) {
  int i = blockIdx.x * blockDim.x + threadIdx.x;
  if (i < n) p[i] = 0.0f;
}

// ---------------- per-(k, channel) sum / sumsq over nodes ----------------
__global__ void stats_kernel(const float* __restrict__ Z, float* __restrict__ stats, int M) {
  const int e  = threadIdx.x;        // blockDim.x == DIM
  const int k  = blockIdx.y;
  const int r0 = blockIdx.x * 256;
  const int r1 = min(r0 + 256, M);
  const float* zp = Z + (long long)k * M * DIM;
  float acc = 0.0f, acc2 = 0.0f;
  for (int r = r0; r < r1; ++r) {
    float v = zp[(long long)r * DIM + e];
    acc += v; acc2 += v * v;
  }
  atomicAdd(stats + (long long)k * 2 * DIM + e, acc);
  atomicAdd(stats + (long long)k * 2 * DIM + DIM + e, acc2);
}

// ------- normalize + relu + weighted combine over k; writes f32 + bf16 mirror -------
__global__ void combine_kernel(const float* __restrict__ Z, const float* __restrict__ stats,
                               const float* __restrict__ wvec, const float* __restrict__ g,
                               const float* __restrict__ beta, float* __restrict__ out,
                               __bf16* __restrict__ outB, int M, int addOut) {
  long long i = (long long)blockIdx.x * blockDim.x + threadIdx.x;
  if (i >= (long long)M * DIM) return;
  const int e = (int)i & (DIM - 1);
  const float invM = 1.0f / (float)M;
  float acc = addOut ? out[i] : 0.0f;
#pragma unroll
  for (int k = 0; k < 4; ++k) {
    const float sum = stats[k * 2 * DIM + e];
    const float sq  = stats[k * 2 * DIM + DIM + e];
    const float mu  = sum * invM;
    const float var = sq * invM - mu * mu;
    const float z   = Z[(long long)k * M * DIM + i];
    const float zn  = (z - mu) * rsqrtf(var + EPSV) * g[k * DIM + e] + beta[k * DIM + e];
    acc += wvec[k] * fmaxf(zn, 0.0f);
  }
  out[i]  = acc;
  outB[i] = (__bf16)acc;
}

// ---------------- final batchnorm + relu + residual ----------------
__global__ void final_kernel(const float* __restrict__ pre, const float* __restrict__ stats,
                             const float* __restrict__ h, const float* __restrict__ bg,
                             const float* __restrict__ bb, float* __restrict__ out, int M) {
  long long i = (long long)blockIdx.x * blockDim.x + threadIdx.x;
  if (i >= (long long)M * DIM) return;
  const int e = (int)i & (DIM - 1);
  const float invM = 1.0f / (float)M;
  const float mu  = stats[e] * invM;
  const float var = stats[DIM + e] * invM - mu * mu;
  const float v   = (pre[i] - mu) * rsqrtf(var + EPSV) * bg[e] + bb[e];
  out[i] = h[i] + fmaxf(v, 0.0f);
}

// ====================================================================
extern "C" void kernel_launch(void* const* d_in, const int* in_sizes, int n_in,
                              void* d_out, int out_size, void* d_ws, size_t ws_size,
                              hipStream_t stream) {
  (void)in_sizes; (void)n_in; (void)out_size; (void)ws_size;
  const float* h   = (const float*)d_in[0];
  const int*   src = (const int*)d_in[1];
  const int*   dst = (const int*)d_in[2];
  const float* w_f = (const float*)d_in[3];
  const float* w_m = (const float*)d_in[4];
  const float* w_l = (const float*)d_in[5];
  const float* pfW = (const float*)d_in[6];
  const float* pfG = (const float*)d_in[8];
  const float* pfB = (const float*)d_in[9];
  const float* pmW = (const float*)d_in[10];
  const float* pmG = (const float*)d_in[12];
  const float* pmB = (const float*)d_in[13];
  const float* plW = (const float*)d_in[14];
  const float* plG = (const float*)d_in[16];
  const float* plB = (const float*)d_in[17];
  const float* catW = (const float*)d_in[18];
  const float* bng  = (const float*)d_in[20];
  const float* bnb  = (const float*)d_in[21];
  float* out = (float*)d_out;

  const long long P = (long long)NNODES * DIM;   // elements per plane
  char* wp = (char*)d_ws;
  auto allocF = [&](long long n) { float*  q = (float*)wp;  wp += n * 4; return q; };
  auto allocH = [&](long long n) { __bf16* q = (__bf16*)wp; wp += n * 2; return q; };

  float* s1 = allocF(P);  float* s2 = allocF(P);
  float* t0 = allocF(P);  float* t1 = allocF(P);
  float* u0 = allocF(P);  float* u1 = allocF(P);
  float* sP = allocF(P);                 // aggregate sum; reused as out_pre at the end
  int*   xPb = (int*)allocF(P);          // aggregate max bits
  float* Zb = allocF(4 * P);             // 4 z planes
  float* deg   = allocF(NNODES);
  float* stats = allocF(8 * DIM);
  __bf16* Wt = allocH(4LL * DIM * DIM);  // transposed weights, bf16
  __bf16* hB  = allocH(P);
  __bf16* s1B = allocH(P); __bf16* s2B = allocH(P);
  __bf16* t0B = allocH(P); __bf16* t1B = allocH(P);
  __bf16* u0B = allocH(P); __bf16* u1B = allocH(P);
  __bf16* sB = allocH(P);  __bf16* mB = allocH(P);  __bf16* xB = allocH(P);

  const int TOT_BLK  = (int)((P + 255) / 256);                  // plane-wide grids
  const long long SC = (long long)NEDGES * (DIM / 4);
  const int SC_BLK   = (int)((SC + 255) / 256);
  const dim3 GGRID(NNODES / 16, DIM / 256);                     // 16x64 per wave, 4 waves

  auto run_agg = [&](const float* X) {
    agg_init_kernel<<<TOT_BLK, 256, 0, stream>>>(sP, xPb, deg, NNODES);
    scatter_kernel<<<SC_BLK, 256, 0, stream>>>(X, src, dst, sP, xPb, deg);
    agg_fin_kernel<<<TOT_BLK, 256, 0, stream>>>(sP, xPb, deg, sB, mB, xB, NNODES);
  };

  auto run_mixed = [&](const __bf16* XB, const float* W4, const float* G4, const float* B4,
                       const float* wv, float* target, __bf16* targetB, int add) {
    wconv_kernel<<<(4 * DIM * DIM + 255) / 256, 256, 0, stream>>>(W4, Wt, 4);
    const __bf16* Aptr[4] = {XB, sB, mB, xB};
    for (int k = 0; k < 4; ++k)
      gemm_kernel<<<GGRID, 128, 0, stream>>>(
          Aptr[k], Wt + (long long)k * DIM * DIM, Zb + (long long)k * P, NNODES, 0);
    zero_kernel<<<(8 * DIM + 255) / 256, 256, 0, stream>>>(stats, 8 * DIM);
    stats_kernel<<<dim3((NNODES + 255) / 256, 4), DIM, 0, stream>>>(Zb, stats, NNODES);
    combine_kernel<<<TOT_BLK, 256, 0, stream>>>(Zb, stats, wv, G4, B4,
                                                target, targetB, NNODES, add);
  };

  const long long WM = 4LL * DIM * DIM;  // floats per mixed-W block
  const int       GM = 4 * DIM;          // floats per mixed-g/beta block

  cvt_bf16_kernel<<<TOT_BLK, 256, 0, stream>>>(h, hB, P);

  // group 1: aggregate(h) -> s1 = mixed(wf0,h); s2 = mixed(wf1,h)
  run_agg(h);
  run_mixed(hB, pfW + 0 * WM, pfG + 0 * GM, pfB + 0 * GM, w_f + 0, s1, s1B, 0);
  run_mixed(hB, pfW + 1 * WM, pfG + 1 * GM, pfB + 1 * GM, w_f + 4, s2, s2B, 0);
  // group 2: aggregate(s1) -> s2 += mixed(wf2,s1); t0 = mixed(wm0,s1)
  run_agg(s1);
  run_mixed(s1B, pfW + 2 * WM, pfG + 2 * GM, pfB + 2 * GM, w_f + 8, s2, s2B, 1);
  run_mixed(s1B, pmW + 0 * WM, pmG + 0 * GM, pmB + 0 * GM, w_m + 0, t0, t0B, 0);
  // group 3: aggregate(s2) -> t1 = mixed(wm1,s2)
  run_agg(s2);
  run_mixed(s2B, pmW + 1 * WM, pmG + 1 * GM, pmB + 1 * GM, w_m + 4, t1, t1B, 0);
  // group 4: aggregate(t0) -> u0 = mixed(wl0,t0); u1 = mixed(wl2,t0)
  run_agg(t0);
  run_mixed(t0B, plW + 0 * WM, plG + 0 * GM, plB + 0 * GM, w_l + 0,  u0, u0B, 0);
  run_mixed(t0B, plW + 2 * WM, plG + 2 * GM, plB + 2 * GM, w_l + 8,  u1, u1B, 0);
  // group 5: aggregate(t1) -> u0 += mixed(wl1,t1); u1 += mixed(wl3,t1)
  run_agg(t1);
  run_mixed(t1B, plW + 1 * WM, plG + 1 * GM, plB + 1 * GM, w_l + 4,  u0, u0B, 1);
  run_mixed(t1B, plW + 3 * WM, plG + 3 * GM, plB + 3 * GM, w_l + 12, u1, u1B, 1);
  // group 6: aggregate(u0) -> u1 += mixed(wl4,u0)
  run_agg(u0);
  run_mixed(u0B, plW + 4 * WM, plG + 4 * GM, plB + 4 * GM, w_l + 16, u1, u1B, 1);

  // final: out_pre = sum_j state_j @ catW[j*512:(j+1)*512, :]  (cat_b cancels in BN)
  wconv_kernel<<<(4 * DIM * DIM + 255) / 256, 256, 0, stream>>>(catW, Wt, 4);
  float* pre = sP;  // aggregate scratch is free now
  const __bf16* Sts[4] = {t0B, t1B, u0B, u1B};
  for (int j = 0; j < 4; ++j)
    gemm_kernel<<<GGRID, 128, 0, stream>>>(
        Sts[j], Wt + (long long)j * DIM * DIM, pre, NNODES, j > 0);
  zero_kernel<<<(2 * DIM + 255) / 256, 256, 0, stream>>>(stats, 2 * DIM);
  stats_kernel<<<dim3((NNODES + 255) / 256, 1), DIM, 0, stream>>>(pre, stats, NNODES);
  final_kernel<<<TOT_BLK, 256, 0, stream>>>(pre, stats, h, bng, bnb, out, NNODES);
}
